// GCNEncoder_2714419331792
// MI455X (gfx1250) — matrix operand order repaired and checked by reading, not compiled, not run on hardware
//
#include <hip/hip_runtime.h>
#include <hip/hip_bf16.h>

// ---------------------------------------------------------------------------
// GCN encoder for MI455X (gfx1250, wave32).
// GEMMs: v_wmma_f32_16x16x32_bf16, one wave = 16 rows x 128 cols (8 tiles,
// A fragment reused across tiles).  B is pre-packed into the exact WMMA
// fragment layout so each fragment is ONE contiguous 32B load per lane.
// Message passing (HBM-bound part): float4 gathers + global_atomic_add_f32.
// BatchNorm biases b1/b2 cancel exactly under BN -> folded away.
// ---------------------------------------------------------------------------

typedef __attribute__((ext_vector_type(16))) __bf16 v16bf;
typedef __attribute__((ext_vector_type(8)))  __bf16 v8bf;
typedef __attribute__((ext_vector_type(8)))  float  v8f;

#define EPS 1e-5f

// ----------------------------- utility kernels -----------------------------

__global__ void k_fill_f32(float* __restrict__ p, float v, long long n) {
  long long i = (long long)blockIdx.x * blockDim.x + threadIdx.x;
  long long stride = (long long)gridDim.x * blockDim.x;
  for (; i < n; i += stride) p[i] = v;
}

__global__ void k_f2bf(const float* __restrict__ in, __bf16* __restrict__ out, long long n) {
  long long i = (long long)blockIdx.x * blockDim.x + threadIdx.x;
  long long stride = (long long)gridDim.x * blockDim.x;
  for (; i < n; i += stride) out[i] = (__bf16)in[i];
}

// ------------------------- gcn_norm (degree, rsqrt) -------------------------

__global__ void k_deg_scatter(const long long* __restrict__ dst,
                              const float* __restrict__ w,
                              float* __restrict__ deg, long long E) {
  long long i = (long long)blockIdx.x * blockDim.x + threadIdx.x;
  long long stride = (long long)gridDim.x * blockDim.x;
  for (; i < E; i += stride) atomicAdd(&deg[dst[i]], w[i]);
}

__global__ void k_dinv(float* __restrict__ deg, int n) {
  int i = blockIdx.x * blockDim.x + threadIdx.x;
  if (i < n) {
    float d = deg[i];
    deg[i] = d > 0.0f ? rsqrtf(d) : 0.0f;   // in-place deg -> d^-1/2
  }
}

// norm[e] = dinv[src]*w*dinv[dst] for e<E ; norm[E+n] = dinv[n]^2 (self loop)
__global__ void k_norm(const long long* __restrict__ src,
                       const long long* __restrict__ dst,
                       const float* __restrict__ w,
                       const float* __restrict__ dinv,
                       float* __restrict__ norm, long long E, int N) {
  long long i = (long long)blockIdx.x * blockDim.x + threadIdx.x;
  long long total = E + (long long)N;
  long long stride = (long long)gridDim.x * blockDim.x;
  for (; i < total; i += stride) {
    if (i < E) norm[i] = dinv[src[i]] * w[i] * dinv[dst[i]];
    else { float d = dinv[i - E]; norm[i] = d * d; }
  }
}

// --------------------- B pre-pack into WMMA fragment layout -----------------
// For the 16-bit 32x16 B fragment, lane L holds K = kt*32 + (L>>4)*8 + {0..7, 16..23}
// at column ct*16 + (L&15).  Packed index:
//   Bp[ (((kt*8 + ct)*32 + lane)*16 + i ] = B[(kt*32 + (lane>>4)*8 + (i&7) + (i>>3)*16)*128 + ct*16 + (lane&15)]
// => each fragment is 32 lanes x 32 contiguous bytes (one b128-pair load/lane).
// N is fixed at 128 (8 column tiles) for all three GEMMs.

__global__ void k_pack_b(const float* __restrict__ B, __bf16* __restrict__ Bp, int K) {
  int idx = blockIdx.x * blockDim.x + threadIdx.x;       // total = K*128
  if (idx >= K * 128) return;
  const int i    = idx & 15;
  const int lane = (idx >> 4) & 31;
  const int ct   = (idx >> 9) & 7;
  const int kt   = idx >> 12;
  const int k = kt * 32 + (lane >> 4) * 8 + (i & 7) + (i >> 3) * 16;
  const int n = ct * 16 + (lane & 15);
  Bp[idx] = (__bf16)B[k * 128 + n];
}

// ------------------------------ WMMA GEMM ----------------------------------
// C[M,128] = A[M,K] (bf16 row-major) * B[K,128] (packed) [+ bias]
// Block = 256 threads = 8 waves = 128 rows.  One wave: 16 rows x 128 cols,
// 8 accumulators, A fragment loaded once per k-step and reused for 8 WMMAs.

__global__ void k_gemm_bf16(const __bf16* __restrict__ A,
                            const __bf16* __restrict__ Bp,
                            float* __restrict__ C,
                            const float* __restrict__ bias,
                            int M, int K) {
  const int lane = threadIdx.x & 31;
  const int wave = threadIdx.x >> 5;
  const int r0 = (blockIdx.x * 8 + wave) * 16;
  if (r0 >= M) return;                    // wave-uniform: EXEC stays all-ones
  const int kg  = lane >> 4;              // 0 or 1 (half-wave)
  const int l15 = lane & 15;

  v8f acc[8];
#pragma unroll
  for (int ct = 0; ct < 8; ++ct) acc[ct] = (v8f){};

  const __bf16* arow = A + (size_t)(r0 + l15) * K;

  for (int k0 = 0; k0 < K; k0 += 32) {
    // A fragment (16-bit 16x32): lane<16 holds K {0..7,16..23}, lane>=16 holds
    // K {8..15,24..31}; two contiguous 16B loads.
    v8bf alo = *(const v8bf*)(arow + k0 + kg * 8);
    v8bf ahi = *(const v8bf*)(arow + k0 + 16 + kg * 8);
    v16bf a;
#pragma unroll
    for (int i = 0; i < 8; ++i) { a[i] = alo[i]; a[8 + i] = ahi[i]; }

    // 8 packed B fragments for this k-tile: one 32B contiguous load each.
    const __bf16* bp = Bp + ((size_t)(k0 >> 5) * 8) * 512 + lane * 16;
#pragma unroll
    for (int ct = 0; ct < 8; ++ct) {
      v16bf b = *(const v16bf*)(bp + ct * 512);
      acc[ct] = __builtin_amdgcn_wmma_f32_16x16x32_bf16(
          /*neg_a=*/false, a, /*neg_b=*/false, b,
          /*c_mod=*/(short)0, acc[ct], /*reuse_a=*/false, /*reuse_b=*/false);
    }
  }

  // C/D layout: VGPR v holds row M = v + 8*kg, col = lane&15 within tile.
#pragma unroll
  for (int ct = 0; ct < 8; ++ct) {
    const int col = ct * 16 + l15;
    const float bv = bias ? bias[col] : 0.0f;
#pragma unroll
    for (int v = 0; v < 8; ++v)
      C[(size_t)(r0 + v + 8 * kg) * 128 + col] = acc[ct][v] + bv;
  }
}

// --------------------------- message passing -------------------------------
// agg[dst] += h[src] * norm ; one wave per edge, lane covers 4 of 128 cols.
// e in [E, E+N) encodes the self loop of node e-E.

__global__ void k_edge_scatter(const long long* __restrict__ src,
                               const long long* __restrict__ dst,
                               const float* __restrict__ norm,
                               const float* __restrict__ h,
                               float* __restrict__ agg,
                               long long E, int N) {
  const int lane = threadIdx.x & 31;
  const int wave = threadIdx.x >> 5;
  const long long total = E + (long long)N;
  long long e = (long long)blockIdx.x * 8 + wave;
  const long long stride = (long long)gridDim.x * 8;
  const int c = lane * 4;
  for (; e < total; e += stride) {
    long long s, d;
    if (e < E) { s = src[e]; d = dst[e]; } else { s = d = e - E; }
    const float nw = norm[e];
    const float4 v = *(const float4*)(h + s * 128 + c);
    float* o = agg + d * 128 + c;
    atomicAdd(o + 0, v.x * nw);
    atomicAdd(o + 1, v.y * nw);
    atomicAdd(o + 2, v.z * nw);
    atomicAdd(o + 3, v.w * nw);
  }
}

// ------------------------------ BatchNorm ----------------------------------

__global__ void k_bn_stats(const float* __restrict__ agg,
                           float* __restrict__ sum, float* __restrict__ sumsq,
                           int N) {
  const int c = threadIdx.x;                 // 128 columns
  float s = 0.f, q = 0.f;
  for (int r = blockIdx.x; r < N; r += gridDim.x) {
    const float v = agg[(size_t)r * 128 + c];
    s += v; q += v * v;
  }
  atomicAdd(&sum[c], s);
  atomicAdd(&sumsq[c], q);
}

__global__ void k_bn_scale(const float* __restrict__ sum,
                           const float* __restrict__ sumsq,
                           const float* __restrict__ gamma,
                           const float* __restrict__ beta,
                           float* __restrict__ sc, float* __restrict__ sh,
                           int N) {
  const int c = threadIdx.x;
  const float inv_n = 1.0f / (float)N;
  const float mean = sum[c] * inv_n;
  const float var  = sumsq[c] * inv_n - mean * mean;
  const float s = gamma[c] * rsqrtf(var + EPS);
  sc[c] = s;
  sh[c] = beta[c] - mean * s;
}

// y = relu(agg*sc + sh) -> bf16 with output row stride ldo (fused BN+ReLU+cvt)
__global__ void k_bn_apply_bf16(const float* __restrict__ agg,
                                const float* __restrict__ sc,
                                const float* __restrict__ sh,
                                __bf16* __restrict__ out, int ldo, int N) {
  const int c = threadIdx.x;
  const float s = sc[c], h = sh[c];
  for (int r = blockIdx.x; r < N; r += gridDim.x) {
    const float v = fmaxf(agg[(size_t)r * 128 + c] * s + h, 0.0f);
    out[(size_t)r * ldo + c] = (__bf16)v;
  }
}

// out[r, coloff+c] = relu(feat[r]*W[c] + b[c])  (rank-1 feature expansion)
__global__ void k_feat_bf16(const float* __restrict__ feat,
                            const float* __restrict__ W,
                            const float* __restrict__ b,
                            __bf16* __restrict__ out, int ldo, int coloff,
                            int N) {
  const int c = threadIdx.x;
  const float w = W[c], bb = b[c];
  for (int r = blockIdx.x; r < N; r += gridDim.x)
    out[(size_t)r * ldo + coloff + c] = (__bf16)fmaxf(feat[r] * w + bb, 0.0f);
}

// ------------------------------- launcher ----------------------------------

extern "C" void kernel_launch(void* const* d_in, const int* in_sizes, int n_in,
                              void* d_out, int out_size, void* d_ws, size_t ws_size,
                              hipStream_t stream) {
  const float*     x       = (const float*)d_in[0];
  const long long* ei      = (const long long*)d_in[1];   // [2, E] int64
  const float*     ew      = (const float*)d_in[2];
  const float*     distf   = (const float*)d_in[3];
  const float*     degf    = (const float*)d_in[4];
  const float*     W1      = (const float*)d_in[5];
  // b1 (d_in[6]) provably cancels under BatchNorm -> unused
  const float*     W2      = (const float*)d_in[7];
  // b2 (d_in[8]) provably cancels under BatchNorm -> unused
  const float*     g1      = (const float*)d_in[9];
  const float*     be1     = (const float*)d_in[10];
  const float*     g2      = (const float*)d_in[11];
  const float*     be2     = (const float*)d_in[12];
  const float*     Wd      = (const float*)d_in[13];
  const float*     bd      = (const float*)d_in[14];
  const float*     Wg      = (const float*)d_in[15];
  const float*     bg      = (const float*)d_in[16];
  const float*     Wm      = (const float*)d_in[17];
  const float*     bm      = (const float*)d_in[18];
  float*           out     = (float*)d_out;

  const int       N  = in_sizes[3];             // 100000 (dist_feat is [N,1])
  const long long E  = in_sizes[2];             // 1600000
  const int       IN = in_sizes[0] / N;         // 64
  const int       H  = 128;                     // hidden = out = 128
  const int       CAT = 3 * H;                  // 384

  const long long* src = ei;
  const long long* dst = ei + E;

  // ---- workspace carve-up (256B aligned) ----
  char* base = (char*)d_ws;
  size_t off = 0;
  auto alloc = [&](size_t bytes) -> char* {
    char* p = base + off;
    off = (off + bytes + 255) & ~(size_t)255;
    return p;
  };
  float*  deg   = (float*)alloc((size_t)N * 4);
  float*  norm  = (float*)alloc((size_t)(E + N) * 4);
  __bf16* xh    = (__bf16*)alloc((size_t)N * IN * 2);
  __bf16* w1p   = (__bf16*)alloc((size_t)IN * H * 2);   // packed W1
  __bf16* w2p   = (__bf16*)alloc((size_t)H * H * 2);    // packed W2
  __bf16* wmp   = (__bf16*)alloc((size_t)CAT * H * 2);  // packed Wm
  float*  t     = (float*)alloc((size_t)N * H * 4);     // GEMM output
  float*  agg   = (float*)alloc((size_t)N * H * 4);     // scatter accumulator
  __bf16* hh    = (__bf16*)alloc((size_t)N * H * 2);    // conv2 input (bf16)
  __bf16* cat   = (__bf16*)alloc((size_t)N * CAT * 2);  // concat (bf16)
  float*  stats = (float*)alloc(4 * 128 * 4);           // ssum|ssq|sc|sh
  float* ssum = stats, *ssq = stats + 128, *scv = stats + 256, *shv = stats + 384;
  (void)ws_size; (void)n_in; (void)out_size;

  const long long ET = E + N;

  // ---- 1) gcn_norm ----
  k_fill_f32<<<2048, 256, 0, stream>>>(deg, 1.0f, N);          // self-loop weight
  k_deg_scatter<<<4096, 256, 0, stream>>>(dst, ew, deg, E);
  k_dinv<<<(N + 255) / 256, 256, 0, stream>>>(deg, N);
  k_norm<<<4096, 256, 0, stream>>>(src, dst, ew, deg, norm, E, N);

  // ---- 2) conversions / weight packing ----
  k_f2bf<<<4096, 256, 0, stream>>>(x, xh, (long long)N * IN);
  k_pack_b<<<(IN * H + 255) / 256, 256, 0, stream>>>(W1, w1p, IN);
  k_pack_b<<<(H * H + 255) / 256, 256, 0, stream>>>(W2, w2p, H);
  k_pack_b<<<(CAT * H + 255) / 256, 256, 0, stream>>>(Wm, wmp, CAT);

  const unsigned gemm_blocks = (unsigned)((N + 127) / 128);

  // ---- 3) conv1: t = xh @ W1 ; agg = scatter(t) ; BN+ReLU -> hh ----
  k_gemm_bf16<<<gemm_blocks, 256, 0, stream>>>(xh, w1p, t, nullptr, N, IN);
  k_fill_f32<<<4096, 256, 0, stream>>>(agg, 0.0f, (long long)N * H);
  k_edge_scatter<<<(unsigned)((ET + 7) / 8), 256, 0, stream>>>(src, dst, norm, t, agg, E, N);
  k_fill_f32<<<1, 256, 0, stream>>>(ssum, 0.0f, 256);
  k_bn_stats<<<1024, 128, 0, stream>>>(agg, ssum, ssq, N);
  k_bn_scale<<<1, 128, 0, stream>>>(ssum, ssq, g1, be1, scv, shv, N);
  k_bn_apply_bf16<<<2048, 128, 0, stream>>>(agg, scv, shv, hh, H, N);

  // ---- 4) conv2: t = hh @ W2 ; agg = scatter(t) ; BN+ReLU -> cat[:,0:128] ----
  k_gemm_bf16<<<gemm_blocks, 256, 0, stream>>>(hh, w2p, t, nullptr, N, H);
  k_fill_f32<<<4096, 256, 0, stream>>>(agg, 0.0f, (long long)N * H);
  k_edge_scatter<<<(unsigned)((ET + 7) / 8), 256, 0, stream>>>(src, dst, norm, t, agg, E, N);
  k_fill_f32<<<1, 256, 0, stream>>>(ssum, 0.0f, 256);
  k_bn_stats<<<1024, 128, 0, stream>>>(agg, ssum, ssq, N);
  k_bn_scale<<<1, 128, 0, stream>>>(ssum, ssq, g2, be2, scv, shv, N);
  k_bn_apply_bf16<<<2048, 128, 0, stream>>>(agg, scv, shv, cat, CAT, N);

  // ---- 5) rank-1 feature expansions into cat[:,128:256] and cat[:,256:384] ----
  k_feat_bf16<<<2048, 128, 0, stream>>>(distf, Wd, bd, cat, CAT, 128, N);
  k_feat_bf16<<<2048, 128, 0, stream>>>(degf,  Wg, bg, cat, CAT, 256, N);

  // ---- 6) out = cat @ Wm + bm ----
  k_gemm_bf16<<<gemm_blocks, 256, 0, stream>>>(cat, wmp, out, bm, N, CAT);
}
